// ESN_58566174049062
// MI455X (gfx1250) — matrix operand order repaired and checked by reading, not compiled, not run on hardware
//
#include <hip/hip_runtime.h>
#include <hip/hip_bf16.h>

// ---------------- problem constants ----------------
#define ESN_B   256
#define ESN_T   1000
#define ESN_F   64
#define ESN_H   1024
#define ESN_K   (ESN_F + ESN_H)   // 1088 fused K
#define ESN_O   10
#define ESN_N   8                 // last_n_outputs
#define ESN_LEAK 0.3f

typedef __attribute__((ext_vector_type(16))) __bf16 bf16x16;
typedef __attribute__((ext_vector_type(8)))  __bf16 bf16x8;
typedef __attribute__((ext_vector_type(8)))  float  f32x8;
typedef __attribute__((ext_vector_type(4)))  int    v4i;

#if defined(__has_builtin)
#if __has_builtin(__builtin_amdgcn_global_load_async_to_lds_b128)
#define HAVE_ASYNC_COPY 1
#endif
#endif

__device__ __forceinline__ void wait_async0() {
#if defined(HAVE_ASYNC_COPY)
#if __has_builtin(__builtin_amdgcn_s_wait_asynccnt)
    __builtin_amdgcn_s_wait_asynccnt(0);
#else
    asm volatile("s_wait_asynccnt 0" ::: "memory");
#endif
#endif
}

__device__ __forceinline__ void copy16B_to_lds(const __bf16* g, __bf16* l) {
#if defined(HAVE_ASYNC_COPY)
    __builtin_amdgcn_global_load_async_to_lds_b128((v4i*)g, (v4i*)l, 0, 0);
#else
    *reinterpret_cast<int4*>(l) = *reinterpret_cast<const int4*>(g);
#endif
}

__device__ __forceinline__ bf16x16 combine16(bf16x8 lo, bf16x8 hi) {
    return __builtin_shufflevector(lo, hi, 0, 1, 2, 3, 4, 5, 6, 7,
                                           8, 9, 10, 11, 12, 13, 14, 15);
}

__device__ __forceinline__ __bf16 f2bf(float f) {
    union { float f; unsigned u; } v; v.f = f;
    unsigned r = v.u + 0x7FFFu + ((v.u >> 16) & 1u);   // round-to-nearest-even
    unsigned short s = (unsigned short)(r >> 16);
    __bf16 o;
    __builtin_memcpy(&o, &s, 2);
    return o;
}

// Convert [w_in ; w_r] (f32, row-major) -> fused bf16 weight matrix [1088 x 1024]
__global__ void esn_prep(const float* __restrict__ w_in,
                         const float* __restrict__ w_r,
                         __bf16* __restrict__ W) {
    int i = blockIdx.x * blockDim.x + threadIdx.x;
    const int nIn = ESN_F * ESN_H;               // 65536
    const int nTot = ESN_K * ESN_H;              // 1114112
    if (i < nTot) {
        float v = (i < nIn) ? w_in[i] : w_r[i - nIn];
        W[i] = f2bf(v);
    }
}

// One recurrence step: h_out = (1-leak)*h_in + leak*tanh([x_t | h_in] @ W)
// GEMM: M=256, N=1024, K=k_end (64 for t==0 else 1088), bf16 WMMA 16x16x32.
// Block = 256 threads = 8 wave32s; block tile 64(M) x 128(N); wave tile 16 x 64.
// B operand staged through double-buffered LDS via async global->LDS copies.
#define BT_PAD 8                                  // +16B row pad (bank spread)
#define BT_LDW (128 + BT_PAD)
__global__ void __launch_bounds__(256)
esn_step(const float*  __restrict__ x,          // [B, T, F] f32
         const __bf16* __restrict__ W,          // [1088, 1024] bf16
         const __bf16* __restrict__ h_in_bf16,  // [B, H]
         const float*  __restrict__ h_in_f32,   // [B, H]
         __bf16*       __restrict__ h_out_bf16, // [B, H]
         float*        __restrict__ h_out_f32,  // [B, H]
         float*        __restrict__ hs,         // [8, B, H] history
         int t, int k_end, int first, int hist_idx) {
    __shared__ __align__(16) __bf16 Bt[2][32][BT_LDW];

    const int tid    = threadIdx.x;
    const int lane   = tid & 31;
    const int wave   = tid >> 5;                // 0..7
    const int wave_m = wave & 3;                // 4 M-subtiles
    const int wave_n = wave >> 2;               // 2 N-subtiles
    const int m0  = blockIdx.y * 64 + wave_m * 16;
    const int nb0 = blockIdx.x * 128;           // block N origin
    const int n0  = nb0 + wave_n * 64;          // wave N origin

    // ---- B tile staging: 32(K) x 128(N) bf16, 32B per thread (2 x b128) ----
    const int cp_k = tid >> 3;                  // 0..31
    const int cp_n = (tid & 7) << 4;            // 0,16,...,112
    auto stage = [&](int k0, int buf) {
        const __bf16* g = W + (size_t)(k0 + cp_k) * ESN_H + nb0 + cp_n;
        __bf16* l = &Bt[buf][cp_k][cp_n];
        copy16B_to_lds(g, l);
        copy16B_to_lds(g + 8, l + 8);
    };

    // A-fragment lane mapping (16-bit A, 16x32): lanes 0-15 -> K {0..7,16..23},
    // lanes 16-31 -> K {8..15,24..31}; row M = lane & 15.
    const int a_row  = m0 + (lane & 15);
    const int a_koff = (lane >> 4) << 3;        // 0 or 8

    f32x8 c0 = {}, c1 = {}, c2 = {}, c3 = {};
    const int nk = k_end >> 5;                  // 2 (t==0) or 34

    // A loaders: x slice (f32 -> bf16) used for the first two K-iterations,
    // bf16 h state for the steady-state loop.
    auto loadA_x = [&](int k0) -> bf16x16 {
        const int kb = k0 + a_koff;
        const float* p = x + (size_t)a_row * (ESN_T * ESN_F)
                           + (size_t)t * ESN_F + kb;
        bf16x16 a;
#pragma unroll
        for (int i = 0; i < 8; ++i) { a[i] = f2bf(p[i]); a[8 + i] = f2bf(p[16 + i]); }
        return a;
    };
    auto loadA_h = [&](int k0) -> bf16x16 {
        const __bf16* p = h_in_bf16 + (size_t)a_row * ESN_H + (k0 + a_koff - ESN_F);
        return combine16(*reinterpret_cast<const bf16x8*>(p),
                         *reinterpret_cast<const bf16x8*>(p + 16));
    };

    // Pipelined tile body: wait own async copies, barrier, prefetch next tile,
    // then overlap B-fragment LDS reads with WMMA issue.
    auto body = [&](int it, bf16x16 a) {
        const int buf = it & 1;
        wait_async0();          // this wave's tile-`it` transfers complete
        __syncthreads();        // all waves' transfers visible; prev reads done
        if (it + 1 < nk) stage((it + 1) << 5, buf ^ 1);

        const int bn = wave_n * 64 + (lane & 15);
        auto loadB = [&](int j) -> bf16x16 {
            bf16x16 v;
#pragma unroll
            for (int i = 0; i < 8; ++i) {
                v[i]     = Bt[buf][a_koff + i][bn + 16 * j];
                v[8 + i] = Bt[buf][16 + a_koff + i][bn + 16 * j];
            }
            return v;
        };
        bf16x16 b0 = loadB(0);
        bf16x16 b1 = loadB(1);
        c0 = __builtin_amdgcn_wmma_f32_16x16x32_bf16(false, a, false, b0, (short)0, c0, false, false);
        bf16x16 b2 = loadB(2);
        c1 = __builtin_amdgcn_wmma_f32_16x16x32_bf16(false, a, false, b1, (short)0, c1, false, false);
        bf16x16 b3 = loadB(3);
        c2 = __builtin_amdgcn_wmma_f32_16x16x32_bf16(false, a, false, b2, (short)0, c2, false, false);
        c3 = __builtin_amdgcn_wmma_f32_16x16x32_bf16(false, a, false, b3, (short)0, c3, false, false);
    };

    stage(0, 0);
    // peeled x iterations (K = 0..63); these are ALL iterations when t == 0
    body(0, loadA_x(0));
    if (nk > 1) body(1, loadA_x(32));
    // steady-state: pure bf16 h-state path, branch-free A loads
    for (int it = 2; it < nk; ++it) body(it, loadA_h(it << 5));

    // ---- epilogue: tanh + leaky update + state/history stores ----
    // C/D map: lanes 0-15 -> N=lane, M=m0+vgpr; lanes 16-31 -> N=lane-16, M=m0+8+vgpr.
    const int nloc  = lane & 15;
    const int mbase = (lane >> 4) << 3;
    f32x8 cacc[4] = { c0, c1, c2, c3 };
#pragma unroll
    for (int i = 0; i < 4; ++i) {
        const int n = n0 + 16 * i + nloc;
#pragma unroll
        for (int r = 0; r < 8; ++r) {
            const int m = m0 + mbase + r;
            const size_t idx = (size_t)m * ESN_H + n;
            float hh = tanhf(cacc[i][r]);
            float hnew = first ? hh
                               : (1.0f - ESN_LEAK) * h_in_f32[idx] + ESN_LEAK * hh;
            h_out_f32[idx]  = hnew;
            h_out_bf16[idx] = f2bf(hnew);
            if (hist_idx >= 0)
                hs[(size_t)hist_idx * (ESN_B * ESN_H) + idx] = hnew;
        }
    }
}

// out[b, o] = sum_{j,h} hs[j][b][h] * w_lin[o, j*H + h] + b_lin[o]
__global__ void __launch_bounds__(128)
esn_readout(const float* __restrict__ hs,     // [8, B, H]
            const float* __restrict__ w_lin,  // [O, 8*H]
            const float* __restrict__ b_lin,  // [O]
            float* __restrict__ out) {        // [B, O]
    const int b = blockIdx.x;
    const int tid = threadIdx.x;
    float acc[ESN_O];
#pragma unroll
    for (int o = 0; o < ESN_O; ++o) acc[o] = 0.0f;

    for (int idx = tid; idx < ESN_N * ESN_H; idx += 128) {
        const int j = idx >> 10;
        const int h = idx & (ESN_H - 1);
        const float f = hs[(size_t)j * (ESN_B * ESN_H) + (size_t)b * ESN_H + h];
#pragma unroll
        for (int o = 0; o < ESN_O; ++o)
            acc[o] += f * w_lin[(size_t)o * (ESN_N * ESN_H) + idx];
    }

    __shared__ float red[ESN_O][128];
#pragma unroll
    for (int o = 0; o < ESN_O; ++o) red[o][tid] = acc[o];
    __syncthreads();
    for (int s = 64; s > 0; s >>= 1) {
        if (tid < s) {
#pragma unroll
            for (int o = 0; o < ESN_O; ++o) red[o][tid] += red[o][tid + s];
        }
        __syncthreads();
    }
    if (tid < ESN_O) out[b * ESN_O + tid] = red[tid][0] + b_lin[tid];
}

extern "C" void kernel_launch(void* const* d_in, const int* in_sizes, int n_in,
                              void* d_out, int out_size, void* d_ws, size_t ws_size,
                              hipStream_t stream) {
    const float* x     = (const float*)d_in[0];  // [256,1000,64]
    const float* w_in  = (const float*)d_in[1];  // [64,1024]
    const float* w_r   = (const float*)d_in[2];  // [1024,1024]
    const float* w_lin = (const float*)d_in[3];  // [10,8192]
    const float* b_lin = (const float*)d_in[4];  // [10]
    float* out = (float*)d_out;

    // workspace layout (all offsets 256B aligned)
    char* ws = (char*)d_ws;
    const size_t szW  = (size_t)ESN_K * ESN_H * sizeof(__bf16);     // 2,228,224
    const size_t szHf = (size_t)ESN_B * ESN_H * sizeof(float);      // 1,048,576
    const size_t szHb = (size_t)ESN_B * ESN_H * sizeof(__bf16);     //   524,288
    __bf16* Wb = (__bf16*)ws;
    float*  hf = (float*) (ws + szW);                 // 2 buffers
    __bf16* hb = (__bf16*)(ws + szW + 2 * szHf);      // 2 buffers
    float*  hs = (float*) (ws + szW + 2 * szHf + 2 * szHb);  // [8,B,H]
    (void)ws_size; (void)in_sizes; (void)n_in; (void)out_size;

    // 1) one-time weight conversion to bf16 (fused [w_in ; w_r])
    {
        const int n = ESN_K * ESN_H;
        esn_prep<<<(n + 255) / 256, 256, 0, stream>>>(w_in, w_r, Wb);
    }

    // 2) T sequential recurrence steps (graph capture amortizes launches)
    dim3 grid(ESN_H / 128, ESN_B / 64);   // 8 x 4 = 32 workgroups
    dim3 block(256);
    for (int t = 0; t < ESN_T; ++t) {
        const int bin  = (t + 1) & 1;     // buffer written by step t-1
        const int bout = t & 1;
        const int k_end = (t == 0) ? ESN_F : ESN_K;
        const int first = (t == 0) ? 1 : 0;
        const int hist  = (t >= ESN_T - ESN_N) ? (t - (ESN_T - ESN_N)) : -1;
        esn_step<<<grid, block, 0, stream>>>(
            x, Wb,
            hb + (size_t)bin  * ESN_B * ESN_H,
            hf + (size_t)bin  * ESN_B * ESN_H,
            hb + (size_t)bout * ESN_B * ESN_H,
            hf + (size_t)bout * ESN_B * ESN_H,
            hs, t, k_end, first, hist);
    }

    // 3) tiny readout GEMV
    esn_readout<<<ESN_B, 128, 0, stream>>>(hs, w_lin, b_lin, out);
}
